// pva_faster_rcnn_87737591923405
// MI455X (gfx1250) — compile-verified, matrix operand order checked by reading, not run on hardware
//
#include <hip/hip_runtime.h>
#include <hip/hip_bf16.h>

typedef __attribute__((ext_vector_type(16))) _Float16 v16h;
typedef __attribute__((ext_vector_type(8)))  _Float16 v8h;
typedef __attribute__((ext_vector_type(8)))  float    v8f;
typedef __attribute__((ext_vector_type(4)))  float    v4f;

#define RA_PH 7
#define RA_PW 7
#define RA_SR 2
#define RA_SCALE 0.0625f
#define NROI 300
#define NCLS 81
#define CBASE 512
#define RCNN_DIN 4096
#define MTILES 19          // ceil(300/16): all row-tiles kept in one wave's registers

// ---------------------------------------------------------------------------
// WMMA fragment loaders (wave32 16x16x32 f16 layout, CDNA5 ISA 7.12.2):
//   lanes 0-15 : rows/cols 0-15, halves 0-7 = K kb..kb+7, halves 8-15 = kb+16..kb+23
//   lanes 16-31: same rows/cols with kb += 8
// ---------------------------------------------------------------------------

// f16 row-major matrix [rows x lda], vectorized (needs k0 % 8 == 0, lda % 8 == 0)
__device__ __forceinline__ v16h wmma_load_f16(const _Float16* __restrict__ A, int lda,
                                              int row0, int lane, int k0, int rows) {
  int r = row0 + (lane & 15);
  r = (r < rows) ? r : (rows - 1);
  int kb = k0 + ((lane & 16) ? 8 : 0);
  const v8h* p = (const v8h*)(A + (size_t)r * lda + kb);
  v8h lo = p[0];
  v8h hi = p[2];   // +16 halves
  v16h out;
#pragma unroll
  for (int j = 0; j < 8; ++j) { out[j] = lo[j]; out[8 + j] = hi[j]; }
  return out;
}

// fp32 row-major weights [rows x ldw], converted to f16 inline (k0 % 8 == 0)
__device__ __forceinline__ v16h wmma_load_w32(const float* __restrict__ W, int ldw,
                                              int n0, int lane, int k0, int rows) {
  int n = n0 + (lane & 15);
  n = (n < rows) ? n : (rows - 1);
  int kb = k0 + ((lane & 16) ? 8 : 0);
  const v4f* p = (const v4f*)(W + (size_t)n * ldw + kb);
  v4f a0 = p[0], a1 = p[1], b0 = p[4], b1 = p[5];
  v16h out;
#pragma unroll
  for (int j = 0; j < 4; ++j) {
    out[j]      = (_Float16)a0[j];
    out[4 + j]  = (_Float16)a1[j];
    out[8 + j]  = (_Float16)b0[j];
    out[12 + j] = (_Float16)b1[j];
  }
  return out;
}

// ---------------------------------------------------------------------------
// FC GEMM, weights-stream-once design:
//   grid = (N/16 tiles, K-split chunks). Each wave owns one N-tile and a
//   disjoint K-chunk, loops ALL 19 M-tiles in registers (one b-load feeds 19
//   WMMAs), then atomically accumulates fp32 partials.
// ---------------------------------------------------------------------------
__global__ __launch_bounds__(32)
void fc_wmma_kernel(const _Float16* __restrict__ A, const float* __restrict__ W,
                    float* __restrict__ Acc, int M, int N, int K, int kchunk) {
  int lane = threadIdx.x;
  int n0 = blockIdx.x * 16;
  int ks = blockIdx.y * kchunk;
  int ke = ks + kchunk;

  v8f z = {};
  v8f acc[MTILES];
#pragma unroll
  for (int t = 0; t < MTILES; ++t) acc[t] = z;

  int nw = n0 + (lane & 15);
  const float* wrow = W + (size_t)((nw < N) ? nw : (N - 1)) * K;

  for (int k = ks; k < ke; k += 32) {
    __builtin_prefetch(wrow + k + 512, 0, 0);   // keep the HBM weight stream ahead
    v16h b = wmma_load_w32(W, K, n0, lane, k, N);
#pragma unroll
    for (int t = 0; t < MTILES; ++t) {
      v16h a = wmma_load_f16(A, K, t * 16, lane, k, M);
      acc[t] = __builtin_amdgcn_wmma_f32_16x16x32_f16(false, a, false, b,
                                                      (short)0, acc[t], false, false);
    }
  }

  int n = n0 + (lane & 15);
  int mo = (lane & 16) ? 8 : 0;
  if (n < N) {
#pragma unroll
    for (int t = 0; t < MTILES; ++t) {
#pragma unroll
      for (int i = 0; i < 8; ++i) {
        int m = t * 16 + i + mo;
        if (m < M) unsafeAtomicAdd(&Acc[(size_t)m * N + n], acc[t][i]);
      }
    }
  }
}

// ---------------------------------------------------------------------------
// Conv weight repack: fp32 [Cout][Cin][3][3] -> f16 [Cout][kyx][ci], Cin
// zero-padded to Cpad (multiple of 32) so the conv K-loop is divide-free.
// ---------------------------------------------------------------------------
__global__ void conv_w_repack(const float* __restrict__ W, _Float16* __restrict__ Wr,
                              int Cout, int Cin, int Cpad) {
  int idx = blockIdx.x * blockDim.x + threadIdx.x;
  int total = Cout * 9 * Cpad;
  if (idx >= total) return;
  int ci = idx % Cpad; int t = idx / Cpad;
  int kyx = t % 9;     int co = t / 9;
  float v = (ci < Cin) ? W[((size_t)co * Cin + ci) * 9 + kyx] : 0.0f;
  Wr[idx] = (_Float16)v;
}

// ---------------------------------------------------------------------------
// Implicit-GEMM conv 3x3 stride-2 SAME (pad_lo=0, pad_hi=1) + bias + ReLU.
// K order = (kyx, ci) over repacked f16 weights; the im2col B fragment is
// gathered once and reused across MSUB cout-subtiles (4 WMMAs per gather).
// ---------------------------------------------------------------------------
template <int MSUB>
__global__ __launch_bounds__(32)
void conv_wmma_kernel(const float* __restrict__ X, const _Float16* __restrict__ Wr,
                      const float* __restrict__ bias, float* __restrict__ Y,
                      int Cin, int Cpad, int Hin, int Win, int Cout, int OW, int npix) {
  int lane = threadIdx.x;
  int n0 = blockIdx.x * 16;           // output-pixel tile
  int m0 = blockIdx.y * 16 * MSUB;    // cout tile group
  int lda = 9 * Cpad;
  int HW = Hin * Win;

  int nB = n0 + (lane & 15);
  nB = (nB < npix) ? nB : (npix - 1);
  int oy = nB / OW;
  int ox = nB - oy * OW;

  v8f z = {};
  v8f acc[MSUB];
#pragma unroll
  for (int s = 0; s < MSUB; ++s) acc[s] = z;

#pragma unroll
  for (int kyx = 0; kyx < 9; ++kyx) {
    int ky = kyx / 3, kx = kyx - ky * 3;          // constants after unroll
    int iy = oy * 2 + ky;
    int ix = ox * 2 + kx;
    bool valid = (iy < Hin) && (ix < Win);
    const float* xb = X + (size_t)iy * Win + ix;

    for (int ci0 = 0; ci0 < Cpad; ci0 += 32) {
      int kb = ci0 + ((lane & 16) ? 8 : 0);
      v16h b;
#pragma unroll
      for (int j = 0; j < 16; ++j) {
        int ci = kb + ((j < 8) ? j : (8 + j));
        float v = 0.0f;
        if (valid && ci < Cin) v = xb[(size_t)ci * HW];
        b[j] = (_Float16)v;
      }
      int k0 = kyx * Cpad + ci0;
#pragma unroll
      for (int s = 0; s < MSUB; ++s) {
        v16h a = wmma_load_f16(Wr, lda, m0 + s * 16, lane, k0, Cout);
        acc[s] = __builtin_amdgcn_wmma_f32_16x16x32_f16(false, a, false, b,
                                                        (short)0, acc[s], false, false);
      }
    }
  }

  int n = n0 + (lane & 15);
  int mo = (lane & 16) ? 8 : 0;
  if (n < npix) {
#pragma unroll
    for (int s = 0; s < MSUB; ++s) {
#pragma unroll
      for (int i = 0; i < 8; ++i) {
        int co = m0 + s * 16 + i + mo;
        if (co < Cout) {
          float v = acc[s][i] + bias[co];
          Y[(size_t)co * npix + n] = fmaxf(v, 0.0f);
        }
      }
    }
  }
}

// ---------------------------------------------------------------------------
// Zero fill / finalize (bias + optional ReLU, fp32 and/or f16 outputs)
// ---------------------------------------------------------------------------
__global__ void fill_zero_kernel(float* __restrict__ p, int n) {
  int i = blockIdx.x * blockDim.x + threadIdx.x;
  if (i < n) p[i] = 0.0f;
}

template <bool RELU>
__global__ void finalize_kernel(const float* __restrict__ Acc, const float* __restrict__ bias,
                                float* __restrict__ Of, _Float16* __restrict__ Oh,
                                int total, int N) {
  int idx = blockIdx.x * blockDim.x + threadIdx.x;
  if (idx >= total) return;
  int n = idx % N;
  float v = Acc[idx] + bias[n];
  if (RELU) v = fmaxf(v, 0.0f);
  if (Of) Of[idx] = v;
  if (Oh) Oh[idx] = (_Float16)v;
}

// ---------------------------------------------------------------------------
// ROI align: pooled[r, c, py, px] -> f16 fc6 GEMM rows
// ---------------------------------------------------------------------------
__global__ void roi_align_kernel(const float* __restrict__ feat, const float* __restrict__ rois,
                                 _Float16* __restrict__ out, int C, int H, int W, int R) {
  int idx = blockIdx.x * blockDim.x + threadIdx.x;
  int total = R * C * RA_PH * RA_PW;
  if (idx >= total) return;
  int px = idx % RA_PW; int t = idx / RA_PW;
  int py = t % RA_PH;  t /= RA_PH;
  int c = t % C;       int r = t / C;

  const float* roi = rois + r * 5;
  float x1 = roi[1] * RA_SCALE, y1 = roi[2] * RA_SCALE;
  float x2 = roi[3] * RA_SCALE, y2 = roi[4] * RA_SCALE;
  float bw = fmaxf(x2 - x1, 1.0f) / RA_PW;
  float bh = fmaxf(y2 - y1, 1.0f) / RA_PH;
  const float* f = feat + (size_t)c * H * W;

  float acc = 0.0f;
#pragma unroll
  for (int sy = 0; sy < RA_SR; ++sy) {
    float gy = y1 + ((float)(py * RA_SR + sy) + 0.5f) / RA_SR * bh;
    gy = fminf(fmaxf(gy, 0.0f), (float)(H - 1));
    int y0 = (int)floorf(gy);
    int y1i = (y0 + 1 < H) ? y0 + 1 : H - 1;
    float wy = gy - (float)y0;
#pragma unroll
    for (int sx = 0; sx < RA_SR; ++sx) {
      float gx = x1 + ((float)(px * RA_SR + sx) + 0.5f) / RA_SR * bw;
      gx = fminf(fmaxf(gx, 0.0f), (float)(W - 1));
      int x0 = (int)floorf(gx);
      int x1b = (x0 + 1 < W) ? x0 + 1 : W - 1;
      float wx = gx - (float)x0;
      float v = f[y0 * W + x0]   * (1.0f - wy) * (1.0f - wx)
              + f[y0 * W + x1b]  * (1.0f - wy) * wx
              + f[y1i * W + x0]  * wy * (1.0f - wx)
              + f[y1i * W + x1b] * wy * wx;
      acc += v;
    }
  }
  out[(size_t)r * (C * RA_PH * RA_PW) + c * (RA_PH * RA_PW) + py * RA_PW + px] =
      (_Float16)(acc * 0.25f);
}

// ---------------------------------------------------------------------------
// Softmax: one wave32 per row, shfl_xor reductions
// ---------------------------------------------------------------------------
__global__ __launch_bounds__(32)
void softmax_kernel(const float* __restrict__ score, float* __restrict__ prob, int N) {
  int r = blockIdx.x;
  int lane = threadIdx.x;
  const float* s = score + (size_t)r * N;
  float mx = -3.0e38f;
  for (int j = lane; j < N; j += 32) mx = fmaxf(mx, s[j]);
#pragma unroll
  for (int off = 16; off > 0; off >>= 1) mx = fmaxf(mx, __shfl_xor(mx, off, 32));
  float sum = 0.0f;
  for (int j = lane; j < N; j += 32) sum += __expf(s[j] - mx);
#pragma unroll
  for (int off = 16; off > 0; off >>= 1) sum += __shfl_xor(sum, off, 32);
  float inv = 1.0f / sum;
  for (int j = lane; j < N; j += 32) prob[(size_t)r * N + j] = __expf(s[j] - mx) * inv;
}

// ---------------------------------------------------------------------------
extern "C" void kernel_launch(void* const* d_in, const int* in_sizes, int n_in,
                              void* d_out, int out_size, void* d_ws, size_t ws_size,
                              hipStream_t stream) {
  const float* im     = (const float*)d_in[0];
  const float* rois   = (const float*)d_in[1];
  const float* w1     = (const float*)d_in[2];  const float* b1 = (const float*)d_in[3];
  const float* w2     = (const float*)d_in[4];  const float* b2 = (const float*)d_in[5];
  const float* w3     = (const float*)d_in[6];  const float* b3 = (const float*)d_in[7];
  const float* w4     = (const float*)d_in[8];  const float* b4 = (const float*)d_in[9];
  const float* fc6_w  = (const float*)d_in[10]; const float* fc6_b = (const float*)d_in[11];
  const float* fc7_w  = (const float*)d_in[12]; const float* fc7_b = (const float*)d_in[13];
  const float* cls_w  = (const float*)d_in[14]; const float* cls_b = (const float*)d_in[15];
  const float* bbox_w = (const float*)d_in[16]; const float* bbox_b = (const float*)d_in[17];
  float* out = (float*)d_out;

  char* ws = (char*)d_ws;
  size_t off = 0;
  auto alloc = [&](size_t bytes) -> void* {
    void* p = ws + off;
    off += (bytes + 255) & ~(size_t)255;
    return p;
  };
  float*    feat1  = (float*)alloc((size_t)64  * 320 * 320 * 4);
  float*    feat2  = (float*)alloc((size_t)128 * 160 * 160 * 4);
  float*    feat3  = (float*)alloc((size_t)256 * 80  * 80  * 4);
  float*    feat4  = (float*)alloc((size_t)CBASE * 40 * 40 * 4);
  _Float16* wr1    = (_Float16*)alloc((size_t)64  * 9 * 32  * 2);
  _Float16* wr2    = (_Float16*)alloc((size_t)128 * 9 * 64  * 2);
  _Float16* wr3    = (_Float16*)alloc((size_t)256 * 9 * 128 * 2);
  _Float16* wr4    = (_Float16*)alloc((size_t)512 * 9 * 256 * 2);
  _Float16* pooled = (_Float16*)alloc((size_t)NROI * CBASE * 49 * 2);
  float*    acc6   = (float*)alloc((size_t)NROI * RCNN_DIN * 4);
  float*    acc7   = (float*)alloc((size_t)NROI * RCNN_DIN * 4);
  float*    accc   = (float*)alloc((size_t)NROI * NCLS * 4);
  float*    accb   = (float*)alloc((size_t)NROI * 4 * NCLS * 4);
  _Float16* h6     = (_Float16*)alloc((size_t)NROI * RCNN_DIN * 2);
  _Float16* h7     = (_Float16*)alloc((size_t)NROI * RCNN_DIN * 2);
  float*    clssc  = (float*)alloc((size_t)NROI * NCLS * 4);

  dim3 wave(32);
  const int K6 = CBASE * 49;   // 25088

  // ---- conv weight repacks (fp32 -> padded f16, divide-free K order) ----
  conv_w_repack<<<(64 * 9 * 32 + 255) / 256, 256, 0, stream>>>(w1, wr1, 64, 3, 32);
  conv_w_repack<<<(128 * 9 * 64 + 255) / 256, 256, 0, stream>>>(w2, wr2, 128, 64, 64);
  conv_w_repack<<<(256 * 9 * 128 + 255) / 256, 256, 0, stream>>>(w3, wr3, 256, 128, 128);
  conv_w_repack<<<(512 * 9 * 256 + 255) / 256, 256, 0, stream>>>(w4, wr4, 512, 256, 256);

  // ---- conv stack: grid = (pixel tiles, cout groups of 64) ----
  conv_wmma_kernel<4><<<dim3((320 * 320 + 15) / 16, 64 / 64), wave, 0, stream>>>(
      im,    wr1, b1, feat1, 3,   32,  640, 640, 64,  320, 320 * 320);
  conv_wmma_kernel<4><<<dim3((160 * 160 + 15) / 16, 128 / 64), wave, 0, stream>>>(
      feat1, wr2, b2, feat2, 64,  64,  320, 320, 128, 160, 160 * 160);
  conv_wmma_kernel<4><<<dim3((80 * 80 + 15) / 16,   256 / 64), wave, 0, stream>>>(
      feat2, wr3, b3, feat3, 128, 128, 160, 160, 256, 80, 80 * 80);
  conv_wmma_kernel<4><<<dim3((40 * 40 + 15) / 16,   512 / 64), wave, 0, stream>>>(
      feat3, wr4, b4, feat4, 256, 256, 80, 80, CBASE, 40, 40 * 40);

  // ---- ROI align -> f16 GEMM rows ----
  {
    int total = NROI * CBASE * 49;
    roi_align_kernel<<<(total + 255) / 256, 256, 0, stream>>>(
        feat4, rois, pooled, CBASE, 40, 40, NROI);
  }

  // ---- zero fp32 accumulators (graph-safe, re-done every call) ----
  fill_zero_kernel<<<(NROI * RCNN_DIN + 255) / 256, 256, 0, stream>>>(acc6, NROI * RCNN_DIN);
  fill_zero_kernel<<<(NROI * RCNN_DIN + 255) / 256, 256, 0, stream>>>(acc7, NROI * RCNN_DIN);
  fill_zero_kernel<<<(NROI * NCLS + 255) / 256, 256, 0, stream>>>(accc, NROI * NCLS);
  fill_zero_kernel<<<(NROI * 4 * NCLS + 255) / 256, 256, 0, stream>>>(accb, NROI * 4 * NCLS);

  // ---- FC head: weights streamed from HBM exactly once (N-tiles x K-split) ----
  fc_wmma_kernel<<<dim3(RCNN_DIN / 16, 8), wave, 0, stream>>>(
      pooled, fc6_w, acc6, NROI, RCNN_DIN, K6, K6 / 8);               // 3136-K chunks
  finalize_kernel<true><<<(NROI * RCNN_DIN + 255) / 256, 256, 0, stream>>>(
      acc6, fc6_b, nullptr, h6, NROI * RCNN_DIN, RCNN_DIN);

  fc_wmma_kernel<<<dim3(RCNN_DIN / 16, 8), wave, 0, stream>>>(
      h6, fc7_w, acc7, NROI, RCNN_DIN, RCNN_DIN, RCNN_DIN / 8);       // 512-K chunks
  finalize_kernel<true><<<(NROI * RCNN_DIN + 255) / 256, 256, 0, stream>>>(
      acc7, fc7_b, nullptr, h7, NROI * RCNN_DIN, RCNN_DIN);

  fc_wmma_kernel<<<dim3((NCLS + 15) / 16, 16), wave, 0, stream>>>(
      h7, cls_w, accc, NROI, NCLS, RCNN_DIN, RCNN_DIN / 16);          // 256-K chunks
  finalize_kernel<false><<<(NROI * NCLS + 255) / 256, 256, 0, stream>>>(
      accc, cls_b, clssc, nullptr, NROI * NCLS, NCLS);

  fc_wmma_kernel<<<dim3((4 * NCLS + 15) / 16, 16), wave, 0, stream>>>(
      h7, bbox_w, accb, NROI, 4 * NCLS, RCNN_DIN, RCNN_DIN / 16);
  finalize_kernel<false><<<(NROI * 4 * NCLS + 255) / 256, 256, 0, stream>>>(
      accb, bbox_b, out + NROI * 5 + NROI * NCLS, nullptr, NROI * 4 * NCLS, 4 * NCLS);

  // ---- outputs: rois passthrough + softmax(cls_score) ----
  hipMemcpyAsync(out, rois, (size_t)NROI * 5 * sizeof(float),
                 hipMemcpyDeviceToDevice, stream);
  softmax_kernel<<<NROI, 32, 0, stream>>>(clssc, out + NROI * 5, NCLS);
}
